// GraphDecoder_81020263072005
// MI455X (gfx1250) — matrix operand order repaired and checked by reading, not compiled, not run on hardware
//
#include <hip/hip_runtime.h>

// ---------------------------------------------------------------------------
// GraphDecoder fused decode-step for gfx1250 (MI455X), bf16 WMMA pipeline.
//   B=1024, M=32768, H=512, L=128, A=64, NB=4
// Stage 1 (pack kernels): fp32 weights -> bf16 WMMA B-fragments in d_ws.
// Stage 2 (decode kernel): one workgroup (8 wave32) per 32-row tile; each
//   wave owns 2 row-tiles x 4 n-tiles (8 f32 accumulators).  Straight-line
//   per-segment spans; A = ds_load_b128 x2, B = global_load_b128 x2 shared
//   across both row-tiles (2 WMMAs per B fragment), SADDR+byte-offset form.
// ---------------------------------------------------------------------------

typedef __attribute__((ext_vector_type(16))) __bf16 v16bf;
typedef __attribute__((ext_vector_type(8)))  float  v8f;
typedef unsigned int u32;
typedef unsigned short u16;

struct alignas(16) U4 { u32 a, b, c, d; };   // POD 16-byte chunk, 16B aligned
struct F2 { float x, y; };                   // POD 8-byte chunk

union Frag {                                 // one WMMA 16-bit operand (8 VGPRs)
    v16bf bf;
    u32   u[8];
    U4    q[2];
};

// fragment sizes: 32 lanes * 8 dwords = 256 dwords = 1024 bytes per (nt,kt)
constexpr size_t OFF_TOPO_W1 = 0;                                   // 32 nt * 36 kt
constexpr size_t OFF_ATOM_W1 = OFF_TOPO_W1 + (size_t)32 * 36 * 256;
constexpr size_t OFF_BOND_W1 = OFF_ATOM_W1 + (size_t)32 * 36 * 256; // 32 * 52
constexpr size_t OFF_WBOND   = OFF_BOND_W1 + (size_t)32 * 52 * 256; // 32 * 17 (516->544)
constexpr size_t OFF_RBOND   = OFF_WBOND   + (size_t)32 * 17 * 256; // 32 * 18
constexpr size_t OFF_TOPO_W2 = OFF_RBOND   + (size_t)32 * 18 * 256; // 1 * 16 (N 1->16)
constexpr size_t OFF_ATOM_W2 = OFF_TOPO_W2 + (size_t)1  * 16 * 256; // 4 * 16
constexpr size_t OFF_BOND_W2 = OFF_ATOM_W2 + (size_t)4  * 16 * 256; // 1 * 16 (N 4->16)
constexpr size_t WS_DWORDS   = OFF_BOND_W2 + (size_t)1  * 16 * 256;

__device__ __forceinline__ u16 f2bf(float f) {
    u32 u = __float_as_uint(f);
    u32 r = u + 0x7FFFu + ((u >> 16) & 1u);   // round-to-nearest-even
    return (u16)(r >> 16);
}
__device__ __forceinline__ u32 packbf2(float lo, float hi) {
    return (u32)f2bf(lo) | ((u32)f2bf(hi) << 16);
}

// A-operand (16-bit 16x32) per-lane K base for VGPR v: pairs (k0, k0+1)
__device__ __forceinline__ int a_k0(int kh, int v) {
    return (v < 4) ? (kh * 8 + 2 * v) : (16 + kh * 8 + 2 * (v - 4));
}

// A fragment from a 16-dword LDS k-slice: dwords [kh*4..+3] and [8+kh*4..+3].
__device__ __forceinline__ Frag load_a(const u32* slice, int kh) {
    const U4* ap = reinterpret_cast<const U4*>(__builtin_assume_aligned(slice, 16));
    Frag a;
    a.q[0] = ap[kh];
    a.q[1] = ap[2 + kh];
    return a;
}

// A fragment built from a global fp32 row (tiny segments: ctx / one-hots).
__device__ __forceinline__ Frag load_a_gf32(const float* __restrict__ row,
                                            int kbase, int kh, int validK) {
    Frag a;
#pragma unroll
    for (int v = 0; v < 8; ++v) {
        int k0 = kbase + a_k0(kh, v);
        float lo = (k0 < validK)     ? row[k0]     : 0.0f;
        float hi = (k0 + 1 < validK) ? row[k0 + 1] : 0.0f;
        a.u[v] = packbf2(lo, hi);
    }
    return a;
}

// B fragment at BYTE offset `off` from the packed-weight base (SADDR form).
__device__ __forceinline__ Frag load_b(const char* __restrict__ Bc, u32 off) {
    const U4* src = reinterpret_cast<const U4*>(__builtin_assume_aligned(Bc + off, 16));
    Frag b;
    b.q[0] = src[0];
    b.q[1] = src[1];
    return b;
}

__device__ __forceinline__ void init_boff(u32 (&boff)[4], int KT, int wave, int lane) {
#pragma unroll
    for (int j = 0; j < 4; ++j)
        boff[j] = (u32)((wave + 8 * j) * KT) * 1024u + (u32)lane * 32u;   // bytes
}

// One contiguous span of k-tiles with LDS-resident A (both row-tiles).
__device__ __forceinline__ void mma_span(v8f (&c0)[4], v8f (&c1)[4],
                                         const u32* a0, const u32* a1, int nkt,
                                         const char* __restrict__ Bc, u32 (&boff)[4],
                                         int kh) {
    for (int kt = 0; kt < nkt; ++kt) {
        Frag b[4];
#pragma unroll
        for (int j = 0; j < 4; ++j) b[j] = load_b(Bc, boff[j]);
        Frag fa0 = load_a(a0, kh);
        Frag fa1 = load_a(a1, kh);
        __builtin_prefetch(Bc + boff[0] + 2048, 0, 3);   // 2 K-slices ahead
#pragma unroll
        for (int j = 0; j < 4; ++j) {
            c0[j] = __builtin_amdgcn_wmma_f32_16x16x32_bf16(
                false, fa0.bf, false, b[j].bf, (short)0, c0[j], false, false);
            c1[j] = __builtin_amdgcn_wmma_f32_16x16x32_bf16(
                false, fa1.bf, false, b[j].bf, (short)0, c1[j], false, false);
        }
#pragma unroll
        for (int j = 0; j < 4; ++j) boff[j] += 1024u;
        a0 += 16;
        a1 += 16;
    }
}

// One span with global fp32 A rows (ctx gather / one-hot segments).
__device__ __forceinline__ void mma_span_g(v8f (&c0)[4], v8f (&c1)[4],
                                           const float* __restrict__ r0,
                                           const float* __restrict__ r1,
                                           int nkt, int validK,
                                           const char* __restrict__ Bc, u32 (&boff)[4],
                                           int kh) {
    for (int kt = 0; kt < nkt; ++kt) {
        Frag b[4];
#pragma unroll
        for (int j = 0; j < 4; ++j) b[j] = load_b(Bc, boff[j]);
        Frag fa0 = load_a_gf32(r0, kt * 32, kh, validK);
        Frag fa1 = load_a_gf32(r1, kt * 32, kh, validK);
#pragma unroll
        for (int j = 0; j < 4; ++j) {
            c0[j] = __builtin_amdgcn_wmma_f32_16x16x32_bf16(
                false, fa0.bf, false, b[j].bf, (short)0, c0[j], false, false);
            c1[j] = __builtin_amdgcn_wmma_f32_16x16x32_bf16(
                false, fa1.bf, false, b[j].bf, (short)0, c1[j], false, false);
        }
#pragma unroll
        for (int j = 0; j < 4; ++j) boff[j] += 1024u;
    }
}

// Epilogue: +bias, ReLU, bf16 -> 32x512 LDS hidden buffer.
__device__ __forceinline__ void store_hidden(const v8f (&c0)[4], const v8f (&c1)[4],
                                             const float* __restrict__ bias,
                                             u16* __restrict__ hout,
                                             int wave, int lane) {
    const int kh = lane >> 4;
    const int n  = lane & 15;
#pragma unroll
    for (int j = 0; j < 4; ++j) {
        const int col = (wave + 8 * j) * 16 + n;
        const float bv = bias[col];
#pragma unroll
        for (int r = 0; r < 8; ++r) {
            float x0 = c0[j][r] + bv;
            float x1 = c1[j][r] + bv;
            x0 = x0 > 0.0f ? x0 : 0.0f;
            x1 = x1 > 0.0f ? x1 : 0.0f;
            hout[(kh * 8 + r) * 512 + col]        = f2bf(x0);   // rows 0..15
            hout[(16 + kh * 8 + r) * 512 + col]   = f2bf(x1);   // rows 16..31
        }
    }
}

// Second-layer GEMM: A = hidden [32x512] bf16 in LDS, one 16-col N tile.
__device__ __forceinline__ v8f small_gemm(const u32* __restrict__ h32,
                                          const char* __restrict__ Bc,
                                          int nt, int rt, int lane) {
    v8f c = {};
    const int kh = lane >> 4;
    const int mm = lane & 15;
    const u32* a = h32 + (rt * 16 + mm) * 256;
    u32 boff = (u32)nt * 16u * 1024u + (u32)lane * 32u;
    for (int kt = 0; kt < 16; ++kt) {
        Frag fa = load_a(a, kh);
        Frag b = load_b(Bc, boff);
        c = __builtin_amdgcn_wmma_f32_16x16x32_bf16(
            false, fa.bf, false, b.bf, (short)0, c, false, false);
        boff += 1024u;
        a += 16;
    }
    return c;
}

// ---------------------------------------------------------------------------
// Weight packing: W fp32 row-major [K x N] -> B-operand (16-bit 32x16) frags.
// ---------------------------------------------------------------------------
__global__ void pack_w(const float* __restrict__ W, u32* __restrict__ dst,
                       int K, int N, int ktiles, int ntiles) {
    int t = blockIdx.x * 256 + threadIdx.x;
    int total = ntiles * ktiles * 32;
    if (t >= total) return;
    int lane = t & 31;
    int kt = (t >> 5) % ktiles;
    int nt = (t >> 5) / ktiles;
    int n  = lane & 15;
    int kh = lane >> 4;
    int col = nt * 16 + n;
    u32* d = dst + (size_t)t * 8;
#pragma unroll
    for (int v = 0; v < 8; ++v) {
        int k0 = kt * 32 + a_k0(kh, v);
        float lo = (k0 < K && col < N) ? W[(size_t)k0 * N + col] : 0.0f;
        float hi = (k0 + 1 < K && col < N) ? W[(size_t)(k0 + 1) * N + col] : 0.0f;
        d[v] = packbf2(lo, hi);
    }
}

// ---------------------------------------------------------------------------
// Fused decode kernel: one workgroup (8 wave32) per 32-row tile.
// ---------------------------------------------------------------------------
__global__ __launch_bounds__(256, 1) void decode_kernel(
    const float* __restrict__ src,    const float* __restrict__ gvec,
    const float* __restrict__ xnode,  const float* __restrict__ znode,
    const float* __restrict__ atom1h, const float* __restrict__ bond1h,
    const float* __restrict__ topo_b1, const float* __restrict__ topo_b2,
    const float* __restrict__ atom_b1, const float* __restrict__ atom_b2,
    const float* __restrict__ bond_b1, const float* __restrict__ bond_b2,
    const float* __restrict__ rbond_b, const float* __restrict__ wbond_b,
    const int* __restrict__ batch_idx,
    const u32* __restrict__ wp, float* __restrict__ out) {
    // exactly 5 * 32KB = 160 KB of LDS (bf16 activations)
    __shared__ __align__(16) u16 sgvec[32 * 512];
    __shared__ __align__(16) u16 sxh[32 * 512];     // xnode, later re-used as hist
    __shared__ __align__(16) u16 sznode[32 * 512];
    __shared__ __align__(16) u16 scur[32 * 512];
    __shared__ __align__(16) u16 sh[32 * 512];

    const int tid  = threadIdx.x;
    const int wave = tid >> 5;
    const int lane = tid & 31;
    const int row0 = blockIdx.x * 32;
    const int m    = lane & 15;
    const int kh   = lane >> 4;

    u32* g32 = reinterpret_cast<u32*>(sgvec);
    u32* x32 = reinterpret_cast<u32*>(sxh);
    u32* z32 = reinterpret_cast<u32*>(sznode);

    // ---- stage big activations: fp32 global (b64) -> packed bf16 LDS ----
    for (int i = tid; i < 32 * 256; i += 256) {
        int r = i >> 8, kd = i & 255;
        size_t g = (size_t)(row0 + r) * 512 + kd * 2;
        F2 tg = *reinterpret_cast<const F2*>(gvec + g);
        F2 tx = *reinterpret_cast<const F2*>(xnode + g);
        F2 tz = *reinterpret_cast<const F2*>(znode + g);
        g32[i] = packbf2(tg.x, tg.y);
        x32[i] = packbf2(tx.x, tx.y);
        z32[i] = packbf2(tz.x, tz.y);
    }
    __syncthreads();

    const u32* sgvec32  = g32;
    const u32* sxh32    = x32;
    const u32* sznode32 = z32;
    const u32* scur32   = reinterpret_cast<const u32*>(scur);
    const u32* sh32     = reinterpret_cast<const u32*>(sh);

    const char* BpT1 = (const char*)(wp + OFF_TOPO_W1);
    const char* BpA1 = (const char*)(wp + OFF_ATOM_W1);
    const char* BpB1 = (const char*)(wp + OFF_BOND_W1);
    const char* BpW  = (const char*)(wp + OFF_WBOND);
    const char* BpR  = (const char*)(wp + OFF_RBOND);
    const char* BpT2 = (const char*)(wp + OFF_TOPO_W2);
    const char* BpA2 = (const char*)(wp + OFF_ATOM_W2);
    const char* BpB2 = (const char*)(wp + OFF_BOND_W2);

    // per-lane gathered rows for the tiny global-resident segments
    const float* ctx0 = src + (size_t)batch_idx[row0 + m] * 128;
    const float* ctx1 = src + (size_t)batch_idx[row0 + 16 + m] * 128;
    const float* at0  = atom1h + (size_t)(row0 + m) * 64;
    const float* at1  = atom1h + (size_t)(row0 + 16 + m) * 64;
    const float* bd0  = bond1h + (size_t)(row0 + m) * 4;
    const float* bd1  = bond1h + (size_t)(row0 + 16 + m) * 4;

    // ---- topo: h = relu([gvec|xnode|ctx] @ topo_w1 + b1) ----
    {
        v8f c0[4] = {}, c1[4] = {};
        u32 boff[4];
        init_boff(boff, 36, wave, lane);
        mma_span(c0, c1, sgvec32 + m * 256, sgvec32 + (16 + m) * 256, 16, BpT1, boff, kh);
        mma_span(c0, c1, sxh32 + m * 256, sxh32 + (16 + m) * 256, 16, BpT1, boff, kh);
        mma_span_g(c0, c1, ctx0, ctx1, 4, 128, BpT1, boff, kh);
        store_hidden(c0, c1, topo_b1, sh, wave, lane);
    }
    __syncthreads();
    if ((wave & 3) == 0) {            // waves 0,4 -> row-tiles 0,1
        int rt = wave >> 2;
        v8f c = small_gemm(sh32, BpT2, 0, rt, lane);
        if ((lane & 15) == 0) {
            float b2 = topo_b2[0];
#pragma unroll
            for (int r = 0; r < 8; ++r)
                out[(size_t)(row0 + rt * 16 + kh * 8 + r) * 69 + 0] = c[r] + b2;
        }
    }
    __syncthreads();

    // ---- atom: h = relu([gvec|xnode|ctx] @ atom_w1 + b1) ----
    {
        v8f c0[4] = {}, c1[4] = {};
        u32 boff[4];
        init_boff(boff, 36, wave, lane);
        mma_span(c0, c1, sgvec32 + m * 256, sgvec32 + (16 + m) * 256, 16, BpA1, boff, kh);
        mma_span(c0, c1, sxh32 + m * 256, sxh32 + (16 + m) * 256, 16, BpA1, boff, kh);
        mma_span_g(c0, c1, ctx0, ctx1, 4, 128, BpA1, boff, kh);
        store_hidden(c0, c1, atom_b1, sh, wave, lane);
    }
    __syncthreads();
    {                                  // all 8 waves: rt = wave>>2, nt = wave&3
        int rt = wave >> 2;
        int nt = wave & 3;
        v8f c = small_gemm(sh32, BpA2, nt, rt, lane);
        int col = nt * 16 + (lane & 15);
        float b2 = atom_b2[col];
#pragma unroll
        for (int r = 0; r < 8; ++r)
            out[(size_t)(row0 + rt * 16 + kh * 8 + r) * 69 + 1 + col] = c[r] + b2;
    }
    __syncthreads();

    // ---- hist = relu([znode | bond_onehot] @ wbond_w + b), K = 516->544 ----
    {
        v8f c0[4] = {}, c1[4] = {};
        u32 boff[4];
        init_boff(boff, 17, wave, lane);
        mma_span(c0, c1, sznode32 + m * 256, sznode32 + (16 + m) * 256, 16, BpW, boff, kh);
        mma_span_g(c0, c1, bd0, bd1, 1, 4, BpW, boff, kh);
        store_hidden(c0, c1, wbond_b, sxh /*hist*/, wave, lane);
    }
    __syncthreads();

    // ---- cur = relu([hist | atom_onehot] @ rbond_w + b), K = 576 ----
    {
        v8f c0[4] = {}, c1[4] = {};
        u32 boff[4];
        init_boff(boff, 18, wave, lane);
        mma_span(c0, c1, sxh32 + m * 256, sxh32 + (16 + m) * 256, 16, BpR, boff, kh);
        mma_span_g(c0, c1, at0, at1, 2, 64, BpR, boff, kh);
        store_hidden(c0, c1, rbond_b, scur, wave, lane);
    }
    __syncthreads();

    // ---- bond: h = relu([gvec | cur | znode | ctx] @ bond_w1 + b1) ----
    {
        v8f c0[4] = {}, c1[4] = {};
        u32 boff[4];
        init_boff(boff, 52, wave, lane);
        mma_span(c0, c1, sgvec32 + m * 256, sgvec32 + (16 + m) * 256, 16, BpB1, boff, kh);
        mma_span(c0, c1, scur32 + m * 256, scur32 + (16 + m) * 256, 16, BpB1, boff, kh);
        mma_span(c0, c1, sznode32 + m * 256, sznode32 + (16 + m) * 256, 16, BpB1, boff, kh);
        mma_span_g(c0, c1, ctx0, ctx1, 4, 128, BpB1, boff, kh);
        store_hidden(c0, c1, bond_b1, sh, wave, lane);
    }
    __syncthreads();
    if ((wave & 3) == 0) {            // waves 0,4 -> row-tiles 0,1
        int rt = wave >> 2;
        v8f c = small_gemm(sh32, BpB2, 0, rt, lane);
        int n = lane & 15;
        if (n < 4) {
            float b2 = bond_b2[n];
#pragma unroll
            for (int r = 0; r < 8; ++r)
                out[(size_t)(row0 + rt * 16 + kh * 8 + r) * 69 + 65 + n] = c[r] + b2;
        }
    }
}

// ---------------------------------------------------------------------------
extern "C" void kernel_launch(void* const* d_in, const int* in_sizes, int n_in,
                              void* d_out, int out_size, void* d_ws, size_t ws_size,
                              hipStream_t stream) {
    const float* src      = (const float*)d_in[0];
    const float* gvec     = (const float*)d_in[1];
    const float* xnode    = (const float*)d_in[2];
    const float* znode    = (const float*)d_in[3];
    const float* atom1h   = (const float*)d_in[4];
    const float* bond1h   = (const float*)d_in[5];
    const float* topo_w1  = (const float*)d_in[6];
    const float* topo_b1  = (const float*)d_in[7];
    const float* topo_w2  = (const float*)d_in[8];
    const float* topo_b2  = (const float*)d_in[9];
    const float* atom_w1  = (const float*)d_in[10];
    const float* atom_b1  = (const float*)d_in[11];
    const float* atom_w2  = (const float*)d_in[12];
    const float* atom_b2  = (const float*)d_in[13];
    const float* bond_w1  = (const float*)d_in[14];
    const float* bond_b1  = (const float*)d_in[15];
    const float* bond_w2  = (const float*)d_in[16];
    const float* bond_b2  = (const float*)d_in[17];
    const float* rbond_w  = (const float*)d_in[18];
    const float* rbond_b  = (const float*)d_in[19];
    const float* wbond_w  = (const float*)d_in[20];
    const float* wbond_b  = (const float*)d_in[21];
    const int*   batch_idx= (const int*)d_in[22];
    float* out = (float*)d_out;
    u32*   wp  = (u32*)d_ws;

    if (ws_size < WS_DWORDS * sizeof(u32)) return;   // need ~5.1 MB scratch

    auto blocks = [](int threads) { return (threads + 255) / 256; };
    // pack_w(W, dst, K, N, ktiles, ntiles); threads = ntiles*ktiles*32
    pack_w<<<blocks(32 * 36 * 32), 256, 0, stream>>>(topo_w1, wp + OFF_TOPO_W1, 1152, 512, 36, 32);
    pack_w<<<blocks(32 * 36 * 32), 256, 0, stream>>>(atom_w1, wp + OFF_ATOM_W1, 1152, 512, 36, 32);
    pack_w<<<blocks(32 * 52 * 32), 256, 0, stream>>>(bond_w1, wp + OFF_BOND_W1, 1664, 512, 52, 32);
    pack_w<<<blocks(32 * 17 * 32), 256, 0, stream>>>(wbond_w, wp + OFF_WBOND,   516,  512, 17, 32);
    pack_w<<<blocks(32 * 18 * 32), 256, 0, stream>>>(rbond_w, wp + OFF_RBOND,   576,  512, 18, 32);
    pack_w<<<blocks(1  * 16 * 32), 256, 0, stream>>>(topo_w2, wp + OFF_TOPO_W2, 512,  1,   16, 1);
    pack_w<<<blocks(4  * 16 * 32), 256, 0, stream>>>(atom_w2, wp + OFF_ATOM_W2, 512,  64,  16, 4);
    pack_w<<<blocks(1  * 16 * 32), 256, 0, stream>>>(bond_w2, wp + OFF_BOND_W2, 512,  4,   16, 1);

    decode_kernel<<<1024, 256, 0, stream>>>(
        src, gvec, xnode, znode, atom1h, bond1h,
        topo_b1, topo_b2, atom_b1, atom_b2, bond_b1, bond_b2,
        rbond_b, wbond_b, batch_idx, wp, out);
}
